// Sparse3DNA_85959475462199
// MI455X (gfx1250) — compile-verified, matrix operand order checked by reading, not run on hardware
//
#include <hip/hip_runtime.h>
#include <hip/hip_bf16.h>
#include <math.h>
#include <stdint.h>

// ---------------- problem constants (from reference) ----------------
#define F_    8
#define H_    24
#define W_    24
#define N_TOK (F_ * H_ * W_)     // 4608 tokens
#define NSEQ  (N_TOK + 1)        // 4609 rows (BOS + tokens)
#define HEADS 8
#define DH    64
#define DIM   512                // HEADS * DH
#define NKV   28                 // 1 BOS + 27 neighborhood keys
#define SCALEQ 0.125f            // DH^-0.5
#define MPAD  4624               // 289 * 16 (NSEQ rounded up to WMMA M tile)
#define GEMM_K DIM               // all three GEMMs have K = 512

typedef __attribute__((ext_vector_type(16))) __bf16 v16bf;
typedef __attribute__((ext_vector_type(8)))  __bf16 v8bf;
typedef __attribute__((ext_vector_type(8)))  float  v8f;

// ---------------- fp32 -> bf16 convert with row zero-padding ----------------
__global__ void cvt_pad_rows(const float* __restrict__ src, __bf16* __restrict__ dst,
                             int rows_valid, int rows_total, int cols) {
    int i = blockIdx.x * blockDim.x + threadIdx.x;
    int total = rows_total * cols;
    if (i >= total) return;
    int r = i / cols;
    dst[i] = (r < rows_valid) ? (__bf16)src[i] : (__bf16)0.0f;
}

// ---------------- fp32 (K x N, row-major) -> bf16 transposed (N x K) ----------------
__global__ void cvt_transpose(const float* __restrict__ w, __bf16* __restrict__ wT,
                              int K, int N) {
    int i = blockIdx.x * blockDim.x + threadIdx.x;
    if (i >= K * N) return;
    int k = i / N, c = i % N;
    wT[(size_t)c * K + k] = (__bf16)w[i];
}

// ---------------- bf16 WMMA GEMM:  C[M x N] = A[M x K] * Bt[N x K]^T (+bias) ----------------
// Block = 128 threads (4 waves). Block tile: 16 rows x 128 cols.
// The block's whole 16 x K A-strip (16 KB bf16) is staged into LDS once with
// GLOBAL_LOAD_ASYNC_TO_LDS_B128 (ASYNCcnt), then each wave computes TWO 16x16
// N-tiles, sharing one A fragment per WMMA pair (ds_load_b128 from LDS).
// Fragment layout per CDNA5 ISA 7.12.2 (wave32):
//   A  (16x32 bf16): lane L: M = L&15, holds K = kb..kb+7 and kb+16..kb+23, kb = 8*(L>>4)
//   B  (32x16 bf16): lane L: N = L&15, same K striping (Bt pre-transposed -> contiguous)
//   C/D(16x16 f32) : VGPR r: M = r + 8*(L>>4), N = L&15
__global__ __launch_bounds__(128)
void wmma_gemm(const __bf16* __restrict__ A, const __bf16* __restrict__ Bt,
               const float* __restrict__ bias, float* __restrict__ C,
               int N, int Mstore) {
    constexpr int K = GEMM_K;
    __shared__ __bf16 sA[16 * K];                       // 16 KB

    const int lane  = threadIdx.x & 31;
    const int wave  = threadIdx.x >> 5;
    const int mbase = blockIdx.x * 16;
    const int ctile = blockIdx.y * 128 + wave * 32;     // two adjacent 16-col tiles
    const int r16   = lane & 15;
    const int kb    = (lane >> 4) * 8;

    // ---- async-stage the contiguous 16xK A-strip into LDS (16 B per lane per op) ----
    {
        const __bf16* gbase = A + (size_t)mbase * K;    // 16 rows, contiguous 16384 B
        uint32_t lds0 = (uint32_t)(uintptr_t)(&sA[0]);  // low 32 bits = LDS byte offset
        constexpr int nchunk = (16 * K * 2) / (128 * 16);
#pragma unroll
        for (int i = 0; i < nchunk; ++i) {
            uint32_t boff = (uint32_t)(threadIdx.x + i * 128) * 16u;
            uint32_t ldst = lds0 + boff;
            asm volatile("global_load_async_to_lds_b128 %0, %1, %2"
                         :: "v"(ldst), "v"(boff), "s"(gbase)
                         : "memory");
        }
        asm volatile("s_wait_asynccnt 0x0" ::: "memory");
    }
    __syncthreads();

    const __bf16* ap  = &sA[r16 * K + kb];                         // LDS A fragment base
    const __bf16* bp0 = Bt + (size_t)(ctile + r16) * K + kb;       // B tile 0 stream
    const __bf16* bp1 = Bt + (size_t)(ctile + 16 + r16) * K + kb;  // B tile 1 stream

    v8f acc0 = {}, acc1 = {};
    for (int kk = 0; kk < K; kk += 32) {
        v8bf alo = *(const v8bf*)(ap + kk);             // ds_load_b128
        v8bf ahi = *(const v8bf*)(ap + kk + 16);
        v8bf b0l = *(const v8bf*)(bp0 + kk);            // global_load_b128
        v8bf b0h = *(const v8bf*)(bp0 + kk + 16);
        v8bf b1l = *(const v8bf*)(bp1 + kk);
        v8bf b1h = *(const v8bf*)(bp1 + kk + 16);
        __builtin_prefetch(bp0 + kk + 64, 0, 1);        // global_prefetch_b8
        __builtin_prefetch(bp1 + kk + 64, 0, 1);
        v16bf a  = __builtin_shufflevector(alo, ahi, 0,1,2,3,4,5,6,7,8,9,10,11,12,13,14,15);
        v16bf b0 = __builtin_shufflevector(b0l, b0h, 0,1,2,3,4,5,6,7,8,9,10,11,12,13,14,15);
        v16bf b1 = __builtin_shufflevector(b1l, b1h, 0,1,2,3,4,5,6,7,8,9,10,11,12,13,14,15);
        // (neg_a, A, neg_b, B, c_mod, C, reuse_a, reuse_b) - two independent chains
        acc0 = __builtin_amdgcn_wmma_f32_16x16x32_bf16(false, a, false, b0, (short)0, acc0,
                                                       false, false);
        acc1 = __builtin_amdgcn_wmma_f32_16x16x32_bf16(false, a, false, b1, (short)0, acc1,
                                                       false, false);
    }

    const int moff = (lane >> 4) * 8;
    const int cN0  = ctile + r16;
    const int cN1  = ctile + 16 + r16;
    const float bv0 = bias ? bias[cN0] : 0.0f;
    const float bv1 = bias ? bias[cN1] : 0.0f;
#pragma unroll
    for (int r = 0; r < 8; ++r) {
        int m = mbase + moff + r;
        if (m < Mstore) {
            C[(size_t)m * N + cN0] = acc0[r] + bv0;
            C[(size_t)m * N + cN1] = acc1[r] + bv1;
        }
    }
}

// ---------------- neighborhood geometry: key slot j -> kv row ----------------
__device__ __forceinline__ bool key_row(int t, int f, int hh, int ww, int j, int* row) {
    if (j == 0) { *row = 0; return true; }           // BOS, never masked
    int a  = j - 1;
    int kf = f  + a / 9       - 1;
    int kh = hh + (a / 3) % 3 - 1;
    int kw = ww + a % 3       - 1;
    if (kf < 0 || kf >= F_ || kh < 0 || kh >= H_ || kw < 0 || kw >= W_) return false;
    int u = (kf * H_ + kh) * W_ + kw;
    if (u > t) return false;                          // causal: only keys <= query
    *row = u + 1;                                     // +1: row 0 is BOS
    return true;
}

// ---------------- attention: 1 block per token, 8 waves = 8 heads ----------------
__global__ __launch_bounds__(256)
void attn_kernel(const float* __restrict__ q,     // MPAD x 512 (token t -> row t+1)
                 const float* __restrict__ kv,    // MPAD x 1024 (k: 0..511, v: 512..1023)
                 const float* __restrict__ w_talk,// 8 x 8
                 __bf16* __restrict__ ao)         // MPAD x 512 (bf16 attention output)
{
    __shared__ float s_attn[HEADS][32];
    __shared__ float s_mix[HEADS][32];

    const int t    = blockIdx.x;
    const int head = threadIdx.x >> 5;
    const int lane = threadIdx.x & 31;

    const int f  = t / (H_ * W_);
    const int hh = (t / W_) % H_;
    const int ww = t % W_;

    const int d0 = lane, d1 = lane + 32;              // 2 dims per lane, DH = 64
    const float* qp = q + (size_t)(t + 1) * DIM + head * DH;
    const float q0 = qp[d0] * SCALEQ;
    const float q1 = qp[d1] * SCALEQ;

    // --- sim_j = q . k_j  (wave-wide dot, butterfly reduce; lane j keeps sim_j) ---
    float mysim = -3.0e38f;
    for (int j = 0; j < NKV; ++j) {
        int row; bool valid = key_row(t, f, hh, ww, j, &row);   // wave-uniform
        float p = 0.0f;
        if (valid) {
            const float* kp = kv + (size_t)row * (2 * DIM) + head * DH;
            p = q0 * kp[d0] + q1 * kp[d1];
        }
        for (int off = 16; off > 0; off >>= 1) p += __shfl_xor(p, off, 32);
        if (lane == j) mysim = valid ? p : -3.0e38f;
    }

    // --- softmax over 28 slots (lanes 28..31 carry -inf -> weight 0) ---
    float mx = mysim;
    for (int off = 16; off > 0; off >>= 1) mx = fmaxf(mx, __shfl_xor(mx, off, 32));
    float e = (lane < NKV) ? __expf(mysim - mx) : 0.0f;
    float s = e;
    for (int off = 16; off > 0; off >>= 1) s += __shfl_xor(s, off, 32);
    s_attn[head][lane] = e / s;
    __syncthreads();

    // --- talking heads: head g mixes all heads' attn at slot j = lane ---
    float mix = 0.0f;
#pragma unroll
    for (int h = 0; h < HEADS; ++h)
        mix += w_talk[head * HEADS + h] * s_attn[h][lane];
    s_mix[head][lane] = mix;
    __syncthreads();

    // --- out = sum_j mixed_j * v_j ---
    float o0 = 0.0f, o1 = 0.0f;
    for (int j = 0; j < NKV; ++j) {
        int row;
        if (!key_row(t, f, hh, ww, j, &row)) continue;          // wave-uniform
        const float wj = s_mix[head][j];
        const float* vp = kv + (size_t)row * (2 * DIM) + DIM + head * DH;
        o0 += wj * vp[d0];
        o1 += wj * vp[d1];
    }
    ao[(size_t)(t + 1) * DIM + head * DH + d0] = (__bf16)o0;
    ao[(size_t)(t + 1) * DIM + head * DH + d1] = (__bf16)o1;

    if (t == 0) {
        // BOS output row = v_bos (per-head concat), plus zero the M-padding rows
        const float* vb = kv + DIM + head * DH;
        ao[head * DH + d0] = (__bf16)vb[d0];
        ao[head * DH + d1] = (__bf16)vb[d1];
        for (int i = threadIdx.x; i < (MPAD - NSEQ) * DIM; i += blockDim.x)
            ao[(size_t)NSEQ * DIM + i] = (__bf16)0.0f;
    }
}

// ---------------- host launcher ----------------
extern "C" void kernel_launch(void* const* d_in, const int* in_sizes, int n_in,
                              void* d_out, int out_size, void* d_ws, size_t ws_size,
                              hipStream_t stream) {
    const float* x      = (const float*)d_in[0];   // (1, 4609, 512)
    const float* w_q    = (const float*)d_in[1];   // (512, 512)
    const float* w_kv   = (const float*)d_in[2];   // (512, 1024)
    const float* w_talk = (const float*)d_in[3];   // (8, 8)
    const float* w_out  = (const float*)d_in[4];   // (512, 512)
    const float* b_out  = (const float*)d_in[5];   // (512,)
    float* out = (float*)d_out;                    // (1, 4609, 512) fp32

    char* ws = (char*)d_ws;
    size_t off = 0;
    auto carve = [&](size_t bytes) {
        char* p = ws + off;
        off += (bytes + 255) & ~(size_t)255;
        return (void*)p;
    };
    __bf16* xb   = (__bf16*)carve((size_t)MPAD * DIM * 2);       // x in bf16, M-padded
    __bf16* wqT  = (__bf16*)carve((size_t)DIM * DIM * 2);        // w_q^T
    __bf16* wkvT = (__bf16*)carve((size_t)(2 * DIM) * DIM * 2);  // w_kv^T
    __bf16* woT  = (__bf16*)carve((size_t)DIM * DIM * 2);        // w_out^T
    float*  qf   = (float*) carve((size_t)MPAD * DIM * 4);       // Q (fp32)
    float*  kvf  = (float*) carve((size_t)MPAD * 2 * DIM * 4);   // K|V (fp32)
    __bf16* ao   = (__bf16*)carve((size_t)MPAD * DIM * 2);       // attn output (bf16)
    (void)ws_size; (void)in_sizes; (void)n_in; (void)out_size;

    // 1) precision staging: bf16 activations + pre-transposed bf16 weights
    cvt_pad_rows<<<(MPAD * DIM + 255) / 256, 256, 0, stream>>>(x, xb, NSEQ, MPAD, DIM);
    cvt_transpose<<<(DIM * DIM + 255) / 256, 256, 0, stream>>>(w_q, wqT, DIM, DIM);
    cvt_transpose<<<(DIM * 2 * DIM + 255) / 256, 256, 0, stream>>>(w_kv, wkvT, DIM, 2 * DIM);
    cvt_transpose<<<(DIM * DIM + 255) / 256, 256, 0, stream>>>(w_out, woT, DIM, DIM);

    // 2) Q and KV projections (WMMA bf16 -> fp32)
    dim3 gq(MPAD / 16, DIM / 128);
    wmma_gemm<<<gq, 128, 0, stream>>>(xb, wqT, nullptr, qf, DIM, MPAD);
    dim3 gkv(MPAD / 16, (2 * DIM) / 128);
    wmma_gemm<<<gkv, 128, 0, stream>>>(xb, wkvT, nullptr, kvf, 2 * DIM, MPAD);

    // 3) sparse neighborhood attention + talking heads (wave32 VALU/LDS path)
    attn_kernel<<<N_TOK, 256, 0, stream>>>(qf, kvf, w_talk, ao);

    // 4) output projection + bias (WMMA bf16 -> fp32, direct to d_out)
    dim3 go(MPAD / 16, DIM / 128);
    wmma_gemm<<<go, 128, 0, stream>>>(ao, woT, b_out, out, DIM, NSEQ);
}
